// MixerModel_47949014893046
// MI455X (gfx1250) — matrix operand order repaired and checked by reading, not compiled, not run on hardware
//
#include <hip/hip_runtime.h>
#include <cstdint>
#include <cstddef>

// Problem dims (compile-time constants from the reference)
#define B_      8
#define L_      1024
#define DMODEL  1024
#define DINNER  2048
#define DSTATE  16
#define DCONV   4
#define DTRANK  64
#define ROWS    (B_ * L_)        // 8192 token rows

typedef __attribute__((ext_vector_type(16))) __bf16 bf16x16;
typedef __attribute__((ext_vector_type(8)))  __bf16 bf16x8;
typedef __attribute__((ext_vector_type(8)))  float  f32x8;

// ---------------------------------------------------------------- helpers
__device__ __forceinline__ __bf16 f2bf(float f) {
  union { float f; uint32_t u; } in; in.f = f;
  uint32_t r = in.u + 0x7fffu + ((in.u >> 16) & 1u);   // round-to-nearest-even
  union { uint16_t u; __bf16 b; } out; out.u = (uint16_t)(r >> 16);
  return out.b;
}

__device__ __forceinline__ float softplusf(float x) {
  return (x > 20.f) ? x : log1pf(__expf(x));
}

// ------------------------------------------------- f32 -> bf16 (with row pad)
__global__ void cvt_pad_kernel(const float* __restrict__ src, __bf16* __restrict__ dst,
                               int srcRows, int cols, int dstRows) {
  int i = blockIdx.x * blockDim.x + threadIdx.x;
  int n = dstRows * cols;
  if (i >= n) return;
  int r = i / cols, c = i - r * cols;
  float v = (r < srcRows) ? src[(size_t)r * cols + c] : 0.f;
  dst[i] = f2bf(v);
}

// --------------------------------------- residual add + LayerNorm (D = 1024)
// first==1: resid = hs ; else resid += hs.  Emits bf16 (for GEMM) and/or f32.
__global__ void add_ln_kernel(const float* __restrict__ hs,
                              float* __restrict__ resid,
                              const float* __restrict__ gam,
                              const float* __restrict__ bet,
                              __bf16* __restrict__ obf,
                              float* __restrict__ of32,
                              int first) {
  const int D = DMODEL;
  int row = blockIdx.x;                 // 8192 rows
  int t   = threadIdx.x;                // 256 threads, 4 elems each
  size_t base = (size_t)row * D;
  float v[4]; float sum = 0.f;
#pragma unroll
  for (int j = 0; j < 4; j++) {
    int c = t * 4 + j;
    float r = hs[base + c];
    if (!first) r += resid[base + c];
    v[j] = r; sum += r;
  }
#pragma unroll
  for (int j = 0; j < 4; j++) resid[base + t * 4 + j] = v[j];

  __shared__ float sd[256];
  sd[t] = sum; __syncthreads();
  for (int s = 128; s > 0; s >>= 1) { if (t < s) sd[t] += sd[t + s]; __syncthreads(); }
  float mu = sd[0] * (1.f / D);
  __syncthreads();
  float sq = 0.f;
#pragma unroll
  for (int j = 0; j < 4; j++) { float d = v[j] - mu; sq += d * d; }
  sd[t] = sq; __syncthreads();
  for (int s = 128; s > 0; s >>= 1) { if (t < s) sd[t] += sd[t + s]; __syncthreads(); }
  float rs = rsqrtf(sd[0] * (1.f / D) + 1e-5f);
#pragma unroll
  for (int j = 0; j < 4; j++) {
    int c = t * 4 + j;
    float o = (v[j] - mu) * rs * gam[c] + bet[c];
    if (obf)  obf[base + c]  = f2bf(o);
    if (of32) of32[base + c] = o;
  }
}

// ------------------------------------------------------ bf16 WMMA GEMM
// C[M,N](f32) = A[M,K](bf16) * W[N,K](bf16)^T  (+bias, optional softplus)
// Wave tile 32x64: 8 accumulators, A-frag reused x4, B-frag reused x2.
// M % 32 == 0, N % 64 == 0, K % 32 == 0 (buffers padded accordingly).
__global__ void __launch_bounds__(256) wmma_gemm_bf16(
    const __bf16* __restrict__ A, const __bf16* __restrict__ W,
    float* __restrict__ C, const float* __restrict__ bias,
    int M, int N, int K, int act) {
  int wave = (int)((blockIdx.x * blockDim.x + threadIdx.x) >> 5);
  int Mt = M >> 5, Nt = N >> 6;
  if (wave >= Mt * Nt) return;
  int mt = wave % Mt, nt = wave / Mt;   // consecutive waves share weights (L2 reuse)
  int m0 = mt << 5, n0 = nt << 6;
  int lane = threadIdx.x & 31;
  int lm = lane & 15, hi = lane >> 4;

  f32x8 zero = {};
  f32x8 acc[2][4];
#pragma unroll
  for (int i = 0; i < 2; i++)
#pragma unroll
    for (int j = 0; j < 4; j++) acc[i][j] = zero;

  // A fragment: lane lm holds row m, elems 0..15 = contiguous K chunk at hi*16
  const __bf16* a0p = A + (size_t)(m0 + lm) * K;
  const __bf16* a1p = A + (size_t)(m0 + 16 + lm) * K;
  // B fragment: lane lm holds col n = weight row n; contiguous K chunk at hi*16
  const __bf16* wp0 = W + (size_t)(n0 +  0 + lm) * K + hi * 16;
  const __bf16* wp1 = W + (size_t)(n0 + 16 + lm) * K + hi * 16;
  const __bf16* wp2 = W + (size_t)(n0 + 32 + lm) * K + hi * 16;
  const __bf16* wp3 = W + (size_t)(n0 + 48 + lm) * K + hi * 16;

  for (int k = 0; k < K; k += 32) {
    if (k + 32 < K) {                       // global_prefetch_b8 for next K-slice
      __builtin_prefetch((const void*)(a0p + k + 32), 0, 1);
      __builtin_prefetch((const void*)(wp0 + k + 32), 0, 1);
    }
    bf16x8 alo0 = *(const bf16x8*)(a0p + k + hi * 16);
    bf16x8 ahi0 = *(const bf16x8*)(a0p + k + hi * 16 + 8);
    bf16x8 alo1 = *(const bf16x8*)(a1p + k + hi * 16);
    bf16x8 ahi1 = *(const bf16x8*)(a1p + k + hi * 16 + 8);
    bf16x16 a0 = __builtin_shufflevector(alo0, ahi0, 0,1,2,3,4,5,6,7,8,9,10,11,12,13,14,15);
    bf16x16 a1 = __builtin_shufflevector(alo1, ahi1, 0,1,2,3,4,5,6,7,8,9,10,11,12,13,14,15);
    bf16x16 b0 = *(const bf16x16*)(wp0 + k);
    bf16x16 b1 = *(const bf16x16*)(wp1 + k);
    bf16x16 b2 = *(const bf16x16*)(wp2 + k);
    bf16x16 b3 = *(const bf16x16*)(wp3 + k);
    acc[0][0] = __builtin_amdgcn_wmma_f32_16x16x32_bf16(false, a0, false, b0, (short)0, acc[0][0], false, false);
    acc[1][0] = __builtin_amdgcn_wmma_f32_16x16x32_bf16(false, a1, false, b0, (short)0, acc[1][0], false, false);
    acc[0][1] = __builtin_amdgcn_wmma_f32_16x16x32_bf16(false, a0, false, b1, (short)0, acc[0][1], false, false);
    acc[1][1] = __builtin_amdgcn_wmma_f32_16x16x32_bf16(false, a1, false, b1, (short)0, acc[1][1], false, false);
    acc[0][2] = __builtin_amdgcn_wmma_f32_16x16x32_bf16(false, a0, false, b2, (short)0, acc[0][2], false, false);
    acc[1][2] = __builtin_amdgcn_wmma_f32_16x16x32_bf16(false, a1, false, b2, (short)0, acc[1][2], false, false);
    acc[0][3] = __builtin_amdgcn_wmma_f32_16x16x32_bf16(false, a0, false, b3, (short)0, acc[0][3], false, false);
    acc[1][3] = __builtin_amdgcn_wmma_f32_16x16x32_bf16(false, a1, false, b3, (short)0, acc[1][3], false, false);
  }

  // D layout: vgpr q, lane -> row = 8*hi + q, col = lm
#pragma unroll
  for (int i = 0; i < 2; i++) {
#pragma unroll
    for (int j = 0; j < 4; j++) {
      int col = n0 + j * 16 + lm;
      int r0  = m0 + i * 16 + 8 * hi;
      float bi = bias ? bias[col] : 0.f;
#pragma unroll
      for (int q = 0; q < 8; q++) {
        float val = acc[i][j][q] + bi;
        if (act) val = softplusf(val);
        C[(size_t)(r0 + q) * N + col] = val;
      }
    }
  }
}

// --------------------------------- causal depthwise conv (k=4) + bias + SiLU
__global__ void conv_silu_kernel(const float* __restrict__ xz,
                                 const float* __restrict__ cw,
                                 const float* __restrict__ cb,
                                 float* __restrict__ xcf,
                                 __bf16* __restrict__ xcbf) {
  int i = blockIdx.x * blockDim.x + threadIdx.x;
  const int n = ROWS * DINNER;
  if (i >= n) return;
  int d  = i & (DINNER - 1);
  int bl = i >> 11;                      // token row (b*L + l)
  int l  = bl & (L_ - 1);
  float acc = cb[d];
#pragma unroll
  for (int j = 0; j < DCONV; j++) {
    int t = l - (DCONV - 1) + j;
    if (t >= 0)
      acc += cw[d * DCONV + j] * xz[(size_t)(bl - l + t) * (2 * DINNER) + d];
  }
  float s = acc / (1.f + __expf(-acc));  // SiLU
  xcf[i]  = s;
  xcbf[i] = f2bf(s);
}

// --------------------------------- dt slice of x_dbl (stride 128) -> bf16
__global__ void dt_cvt_kernel(const float* __restrict__ xdbl, __bf16* __restrict__ dtbf) {
  int i = blockIdx.x * blockDim.x + threadIdx.x;
  const int n = ROWS * DTRANK;
  if (i >= n) return;
  int r = i >> 6, c = i & 63;
  dtbf[i] = f2bf(xdbl[(size_t)r * 128 + c]);
}

// ---------------------------- selective scan: 16 lanes (one per state) / (b,d)
__global__ void scan_kernel(const float* __restrict__ xcf,
                            const float* __restrict__ dtf,
                            const float* __restrict__ xdbl,
                            const float* __restrict__ xz,
                            const float* __restrict__ A_log,
                            const float* __restrict__ Dp,
                            __bf16* __restrict__ ybf) {
  int tid   = blockIdx.x * blockDim.x + threadIdx.x;
  int nlane = tid & 15;                  // state index
  int grp   = tid >> 4;                  // (b, d) channel
  if (grp >= B_ * DINNER) return;
  int d = grp & (DINNER - 1);
  int b = grp >> 11;
  float Av = -__expf(A_log[d * DSTATE + nlane]);
  float Dv = Dp[d];
  float h  = 0.f;
  const float* xrow = xcf  + (size_t)b * L_ * DINNER + d;
  const float* drow = dtf  + (size_t)b * L_ * DINNER + d;
  const float* brow = xdbl + (size_t)b * L_ * 128 + 64 + nlane;   // B slice
  const float* crow = brow + 16;                                  // C slice
  const float* zrow = xz   + (size_t)b * L_ * (2 * DINNER) + DINNER + d;
  __bf16*      yrow = ybf  + (size_t)b * L_ * DINNER + d;
  for (int t = 0; t < L_; t++) {
    float x  = xrow[(size_t)t * DINNER];
    float dt = drow[(size_t)t * DINNER];
    float Bv = brow[(size_t)t * 128];
    float Cv = crow[(size_t)t * 128];
    float dA = __expf(dt * Av);
    h = fmaf(dA, h, dt * x * Bv);
    float y = h * Cv;
    y += __shfl_xor(y, 1);               // butterfly within the 16-lane group
    y += __shfl_xor(y, 2);
    y += __shfl_xor(y, 4);
    y += __shfl_xor(y, 8);
    if (nlane == 0) {
      float z = zrow[(size_t)t * (2 * DINNER)];
      float out = (y + x * Dv) * (z / (1.f + __expf(-z)));   // gate with silu(z)
      yrow[(size_t)t * DINNER] = f2bf(out);
    }
  }
}

// ---------------------------------------------------------------- launcher
extern "C" void kernel_launch(void* const* d_in, const int* in_sizes, int n_in,
                              void* d_out, int out_size, void* d_ws, size_t ws_size,
                              hipStream_t stream) {
  (void)in_sizes; (void)n_in; (void)out_size; (void)ws_size;
  const float* hidden = (const float*)d_in[0];
  const float* in_w   = (const float*)d_in[1];
  const float* cw     = (const float*)d_in[2];
  const float* cb     = (const float*)d_in[3];
  const float* xp_w   = (const float*)d_in[4];
  const float* dtw    = (const float*)d_in[5];
  const float* dtb    = (const float*)d_in[6];
  const float* A_log  = (const float*)d_in[7];
  const float* Dp     = (const float*)d_in[8];
  const float* out_w  = (const float*)d_in[9];
  const float* nw     = (const float*)d_in[10];
  const float* nb     = (const float*)d_in[11];
  const float* nfw    = (const float*)d_in[12];
  const float* nfb    = (const float*)d_in[13];

  char* ws = (char*)d_ws; size_t off = 0;
  auto alloc = [&](size_t bytes) -> void* {
    void* p = ws + off; off += (bytes + 255) & ~(size_t)255; return p;
  };
  float*  resid = (float*) alloc((size_t)ROWS * DMODEL * 4);
  float*  hsbuf = (float*) alloc((size_t)ROWS * DMODEL * 4);
  __bf16* xbf   = (__bf16*)alloc((size_t)ROWS * DMODEL * 2);
  float*  xz    = (float*) alloc((size_t)ROWS * 2 * DINNER * 4);
  float*  xcf   = (float*) alloc((size_t)ROWS * DINNER * 4);
  __bf16* xcbf  = (__bf16*)alloc((size_t)ROWS * DINNER * 2);
  float*  xdbl  = (float*) alloc((size_t)ROWS * 128 * 4);     // padded 96->128
  __bf16* dtbf  = (__bf16*)alloc((size_t)ROWS * DTRANK * 2);
  float*  dtf   = (float*) alloc((size_t)ROWS * DINNER * 4);
  __bf16* ybf   = (__bf16*)alloc((size_t)ROWS * DINNER * 2);
  __bf16* wb_in = (__bf16*)alloc((size_t)4096 * 1024 * 2);
  __bf16* wb_x  = (__bf16*)alloc((size_t)128  * 2048 * 2);
  __bf16* wb_dt = (__bf16*)alloc((size_t)2048 * 64   * 2);
  __bf16* wb_o  = (__bf16*)alloc((size_t)1024 * 2048 * 2);

  auto gemm = [&](const __bf16* Aa, const __bf16* Ww, float* Cc, const float* bias,
                  int M, int N, int K, int act) {
    int waves  = (M >> 5) * (N >> 6);
    int blocks = (waves * 32 + 255) / 256;
    hipLaunchKernelGGL(wmma_gemm_bf16, dim3(blocks), dim3(256), 0, stream,
                       Aa, Ww, Cc, bias, M, N, K, act);
  };
  auto cvt = [&](const float* s, __bf16* dd, int sr, int c, int dr) {
    int n = dr * c;
    hipLaunchKernelGGL(cvt_pad_kernel, dim3((n + 255) / 256), dim3(256), 0, stream,
                       s, dd, sr, c, dr);
  };

  for (int i = 0; i < 4; i++) {
    // per-layer weight copies (bf16, x_proj N padded 96->128 with zeros)
    cvt(in_w  + (size_t)i * 4096 * 1024, wb_in, 4096, 1024, 4096);
    cvt(xp_w  + (size_t)i * 96   * 2048, wb_x,  96,   2048, 128);
    cvt(dtw   + (size_t)i * 2048 * 64,   wb_dt, 2048, 64,   2048);
    cvt(out_w + (size_t)i * 1024 * 2048, wb_o,  1024, 2048, 1024);

    // residual add + LN -> bf16 activations
    hipLaunchKernelGGL(add_ln_kernel, dim3(ROWS), dim3(256), 0, stream,
                       (i == 0 ? hidden : hsbuf), resid,
                       nw + (size_t)i * DMODEL, nb + (size_t)i * DMODEL,
                       xbf, (float*)nullptr, (i == 0) ? 1 : 0);

    // in_proj: (8192x1024)x(4096x1024)^T -> xz (8192x4096)
    gemm(xbf, wb_in, xz, nullptr, ROWS, 2 * DINNER, DMODEL, 0);

    // depthwise causal conv + SiLU -> xc (f32 + bf16)
    {
      int n = ROWS * DINNER;
      hipLaunchKernelGGL(conv_silu_kernel, dim3((n + 255) / 256), dim3(256), 0, stream,
                         xz, cw + (size_t)i * DINNER * DCONV, cb + (size_t)i * DINNER,
                         xcf, xcbf);
    }

    // x_proj: (8192x2048)x(128x2048)^T -> x_dbl (8192x128, last 32 cols padded)
    gemm(xcbf, wb_x, xdbl, nullptr, ROWS, 128, DINNER, 0);

    // dt slice -> bf16
    {
      int n = ROWS * DTRANK;
      hipLaunchKernelGGL(dt_cvt_kernel, dim3((n + 255) / 256), dim3(256), 0, stream,
                         xdbl, dtbf);
    }

    // dt_proj + bias + softplus: (8192x64)x(2048x64)^T -> dt (8192x2048)
    gemm(dtbf, wb_dt, dtf, dtb + (size_t)i * DINNER, ROWS, DINNER, DTRANK, 1);

    // selective scan + D skip + silu(z) gate -> y bf16
    {
      int threads = B_ * DINNER * 16;  // 262144 lanes, 16 states per channel
      hipLaunchKernelGGL(scan_kernel, dim3(threads / 256), dim3(256), 0, stream,
                         xcf, dtf, xdbl, xz,
                         A_log + (size_t)i * DINNER * DSTATE, Dp + (size_t)i * DINNER,
                         ybf);
    }

    // out_proj: (8192x2048)x(1024x2048)^T -> hs (8192x1024)
    gemm(ybf, wb_o, hsbuf, nullptr, ROWS, DMODEL, DINNER, 0);
  }

  // final residual add + LayerNorm -> d_out (f32)
  hipLaunchKernelGGL(add_ln_kernel, dim3(ROWS), dim3(256), 0, stream,
                     hsbuf, resid, nfw, nfb,
                     (__bf16*)nullptr, (float*)d_out, 0);
}
